// BasicTransformer_14688788152624
// MI455X (gfx1250) — matrix operand order repaired and checked by reading, not compiled, not run on hardware
//
#include <hip/hip_runtime.h>
#include <hip/hip_bf16.h>
#include <hip/hip_fp16.h>

// ---------------- model constants ----------------
constexpr int Bc  = 2;
constexpr int Sc  = 2048;
constexpr int Vc  = 50257;
constexpr int Dc  = 1024;
constexpr int Hc  = 16;
constexpr int DHc = 64;
constexpr int Mc  = 4096;
constexpr int Lc  = 4;
constexpr int Tc  = Bc * Sc;          // 4096 tokens

typedef _Float16 h16;
typedef __attribute__((ext_vector_type(16))) _Float16 v16h;
typedef __attribute__((ext_vector_type(8)))  _Float16 v8h;
typedef __attribute__((ext_vector_type(8)))  float    v8f;

#define WMMA16(a, b, c) \
  __builtin_amdgcn_wmma_f32_16x16x32_f16(false, (a), false, (b), (short)0, (c), false, false)

// ---------------- embedding ----------------
__global__ void k_embed(const int* __restrict__ tokens, const float* __restrict__ WE,
                        const float* __restrict__ Wpos, float* __restrict__ resid) {
  int row = blockIdx.x;                 // 0..T-1
  int s   = row % Sc;
  int tok = tokens[row];
  for (int d = threadIdx.x; d < Dc; d += blockDim.x)
    resid[(size_t)row * Dc + d] = WE[(size_t)tok * Dc + d] + Wpos[(size_t)s * Dc + d];
}

// ---------------- layernorm (f32 in -> f16 out) ----------------
__global__ void k_layernorm(const float* __restrict__ x, const float* __restrict__ w,
                            const float* __restrict__ bb, h16* __restrict__ out) {
  __shared__ float red[256];
  int row = blockIdx.x, tid = threadIdx.x;
  const float* xr = x + (size_t)row * Dc;
  float s = 0.f;
  for (int d = tid; d < Dc; d += 256) s += xr[d];
  red[tid] = s; __syncthreads();
  for (int o = 128; o > 0; o >>= 1) { if (tid < o) red[tid] += red[tid + o]; __syncthreads(); }
  float mean = red[0] / (float)Dc;
  __syncthreads();
  float ss = 0.f;
  for (int d = tid; d < Dc; d += 256) { float c = xr[d] - mean; ss += c * c; }
  red[tid] = ss; __syncthreads();
  for (int o = 128; o > 0; o >>= 1) { if (tid < o) red[tid] += red[tid + o]; __syncthreads(); }
  float inv = 1.f / sqrtf(red[0] / (float)Dc + 1e-5f);
  for (int d = tid; d < Dc; d += 256)
    out[(size_t)row * Dc + d] = (h16)(((xr[d] - mean) * inv) * w[d] + bb[d]);
}

// ---------------- transpose + f32->f16 convert ----------------
// in: [batch, R, C] f32 row-major  ->  out: [batch, C, R] f16
__global__ void k_transpose_f32_f16(const float* __restrict__ in, h16* __restrict__ out,
                                    int R, int C, long total) {
  long i = (long)blockIdx.x * blockDim.x + threadIdx.x;
  if (i >= total) return;
  long per = (long)R * C;
  long bz = i / per, rem = i % per;
  int r = (int)(rem / C), c = (int)(rem % C);
  out[bz * per + (long)c * R + r] = (h16)in[i];
}

// ---------------- WMMA GEMM (2x2 register blocking) ----------------
// C[M,N] = A[M,K] (f16, row-major, lda) * Bt[N,K]^T (f16, row-major, ldb)
// + optional bias[N], optional gelu, optional += resid[M,ldr].
// Output: f32 (Cf) if Cf != nullptr, else f16 (Ch).
// Block = 256 thr = 8 waves arranged 2(M) x 4(N); wave tile 32x32 (4 WMMA C tiles);
// block tile 64(M) x 128(N). 4 WMMAs per 4 fragment loads -> 16 FLOP/byte.
__device__ __forceinline__ v16h pack_a(const h16* p, int kk) {
  v8h lo = *(const v8h*)(p + kk);
  v8h hi = *(const v8h*)(p + kk + 16);
  v16h r;
#pragma unroll
  for (int i = 0; i < 8; i++) { r[i] = lo[i]; r[i + 8] = hi[i]; }
  return r;
}
__device__ __forceinline__ v16h pack_b(const h16* p, int kk) {
  v8h lo = *(const v8h*)(p + kk);
  v8h hi = *(const v8h*)(p + kk + 16);
  v16h r;
#pragma unroll
  for (int i = 0; i < 8; i++) { r[i] = lo[i]; r[i + 8] = hi[i]; }
  return r;
}

__global__ void k_gemm_wmma(const h16* __restrict__ A, int lda,
                            const h16* __restrict__ Bt, int ldb,
                            const float* __restrict__ bias,
                            const float* __restrict__ resid, int ldr,
                            float* __restrict__ Cf, h16* __restrict__ Ch, int ldc,
                            int Mr, int N, int K, int flags) {
  int lane = threadIdx.x & 31;
  int wid  = threadIdx.x >> 5;          // 0..7
  int wm   = wid & 1;                   // 2 waves along M
  int wn   = wid >> 1;                  // 4 waves along N
  int row0 = blockIdx.y * 64 + wm * 32;
  int col0 = blockIdx.x * 128 + wn * 32;
  int l16  = lane & 15;
  int half = lane >> 4;                 // 0 or 1

  // clamped fragment source rows (never stored when OOB)
  int ar0 = row0 + l16;      if (ar0 >= Mr) ar0 = Mr - 1;
  int ar1 = row0 + 16 + l16; if (ar1 >= Mr) ar1 = Mr - 1;
  int bc0 = col0 + l16;      if (bc0 >= N)  bc0 = N - 1;
  int bc1 = col0 + 16 + l16; if (bc1 >= N)  bc1 = N - 1;
  const h16* ap0 = A  + (size_t)ar0 * lda + half * 8;
  const h16* ap1 = A  + (size_t)ar1 * lda + half * 8;
  const h16* bp0 = Bt + (size_t)bc0 * ldb + half * 8;
  const h16* bp1 = Bt + (size_t)bc1 * ldb + half * 8;

  v8f acc00 = {}, acc01 = {}, acc10 = {}, acc11 = {};
  for (int kk = 0; kk < K; kk += 32) {
    v16h a0 = pack_a(ap0, kk);
    v16h a1 = pack_a(ap1, kk);
    v16h b0 = pack_b(bp0, kk);
    v16h b1 = pack_b(bp1, kk);
    acc00 = WMMA16(a0, b0, acc00);
    acc01 = WMMA16(a0, b1, acc01);
    acc10 = WMMA16(a1, b0, acc10);
    acc11 = WMMA16(a1, b1, acc11);
  }

  // epilogue: 2 col groups x 2 row groups
  const v8f* accs[4] = { &acc00, &acc01, &acc10, &acc11 };
#pragma unroll
  for (int t = 0; t < 4; t++) {
    int rbase = row0 + (t >> 1) * 16;
    int col   = col0 + (t & 1) * 16 + l16;
    if (col >= N) continue;
    float bv_ = bias ? bias[col] : 0.f;
    v8f acc = *accs[t];
#pragma unroll
    for (int j = 0; j < 8; j++) {
      int row = rbase + j + half * 8;
      if (row < Mr) {
        float vV = acc[j] + bv_;
        if (flags & 1) {   // gelu_new
          float xg = vV;
          vV = 0.5f * xg * (1.f + tanhf(0.7978845608028654f * (xg + 0.044715f * xg * xg * xg)));
        }
        if (flags & 2) vV += resid[(size_t)row * ldr + col];
        if (Cf) Cf[(size_t)row * ldc + col] = vV;
        else    Ch[(size_t)row * ldc + col] = (h16)vV;
      }
    }
  }
}

// ---------------- flash attention (one wave per 16-query tile) ----------------
__device__ inline float hmax16(float v) {
  v = fmaxf(v, __shfl_xor(v, 1, 32));
  v = fmaxf(v, __shfl_xor(v, 2, 32));
  v = fmaxf(v, __shfl_xor(v, 4, 32));
  v = fmaxf(v, __shfl_xor(v, 8, 32));
  return v;
}
__device__ inline float hsum16(float v) {
  v += __shfl_xor(v, 1, 32);
  v += __shfl_xor(v, 2, 32);
  v += __shfl_xor(v, 4, 32);
  v += __shfl_xor(v, 8, 32);
  return v;
}

// q,k,v,z layout: [B, S, H, DH] f16 (row stride H*DH)
__global__ void k_attn(const h16* __restrict__ q, const h16* __restrict__ k,
                       const h16* __restrict__ v, h16* __restrict__ z) {
  __shared__ _Float16 pb[16 * 16];
  int lane = threadIdx.x;
  int qt = blockIdx.x, h = blockIdx.y, b = blockIdx.z;
  int l16 = lane & 15;
  int half = lane >> 4;
  const int rs = Hc * DHc;                               // 1024
  size_t headoff = ((size_t)b * Sc) * rs + (size_t)h * DHc;

  // Q as A-fragments over K=DH (two chunks of 32)
  const h16* qrow = q + headoff + (size_t)(qt * 16 + l16) * rs + (half ? 8 : 0);
  v16h a0, a1;
  {
    v8h x0 = *(const v8h*)(qrow);
    v8h x1 = *(const v8h*)(qrow + 16);
    v8h x2 = *(const v8h*)(qrow + 32);
    v8h x3 = *(const v8h*)(qrow + 48);
#pragma unroll
    for (int i = 0; i < 8; i++) { a0[i] = x0[i]; a0[i + 8] = x1[i]; a1[i] = x2[i]; a1[i + 8] = x3[i]; }
  }

  v8f o0 = {}, o1 = {}, o2 = {}, o3 = {};
  float m_[8], l_[8];
#pragma unroll
  for (int j = 0; j < 8; j++) { m_[j] = -3.0e38f; l_[j] = 0.f; }

  for (int kt = 0; kt <= qt; ++kt) {
    // K-tile as B-fragments (Bt row n = key row, contiguous in dh)
    const h16* krow = k + headoff + (size_t)(kt * 16 + l16) * rs + (half ? 8 : 0);
    v16h b0, b1;
    {
      v8h x0 = *(const v8h*)(krow);
      v8h x1 = *(const v8h*)(krow + 16);
      v8h x2 = *(const v8h*)(krow + 32);
      v8h x3 = *(const v8h*)(krow + 48);
#pragma unroll
      for (int i = 0; i < 8; i++) { b0[i] = x0[i]; b0[i + 8] = x1[i]; b1[i] = x2[i]; b1[i + 8] = x3[i]; }
    }
    v8f sacc = {};
    sacc = WMMA16(a0, b0, sacc);
    sacc = WMMA16(a1, b1, sacc);

    int keyg = kt * 16 + l16;
    float p[8];
#pragma unroll
    for (int j = 0; j < 8; j++) {
      int qg = qt * 16 + j + half * 8;
      float sv = sacc[j] * 0.125f;                 // 1/sqrt(64)
      if (keyg > qg) sv = -100000.0f;              // causal IGNORE
      float mx = hmax16(sv);
      float mn = fmaxf(m_[j], mx);
      float esc = expf(m_[j] - mn);
      float pv = expf(sv - mn);
      l_[j] = l_[j] * esc + hsum16(pv);
      m_[j] = mn;
      p[j] = pv;
      o0[j] *= esc; o1[j] *= esc; o2[j] *= esc; o3[j] *= esc;
    }

    // re-layout P (C-layout) -> A-layout via LDS, zero-pad K 16..31
    __syncthreads();
#pragma unroll
    for (int j = 0; j < 8; j++) pb[(j + half * 8) * 16 + l16] = (h16)p[j];
    __syncthreads();
    v16h pa;
    {
      int off = half * 8;
#pragma unroll
      for (int i = 0; i < 8; i++) { pa[i] = pb[l16 * 16 + off + i]; pa[i + 8] = (h16)0.f; }
    }

    // V-tile as B-fragments [keys x dh], 4 column tiles of 16; K 16..31 zero
    const h16* vbase = v + headoff + (size_t)(kt * 16) * rs;
    v16h vb0, vb1, vb2, vb3;
#pragma unroll
    for (int i = 0; i < 16; i++) {
      h16 e0 = (h16)0.f, e1 = (h16)0.f, e2 = (h16)0.f, e3 = (h16)0.f;
      if (i < 8) {
        const h16* vr = vbase + (size_t)(half * 8 + i) * rs + l16;
        e0 = vr[0]; e1 = vr[16]; e2 = vr[32]; e3 = vr[48];
      }
      vb0[i] = e0; vb1[i] = e1; vb2[i] = e2; vb3[i] = e3;
    }
    o0 = WMMA16(pa, vb0, o0);
    o1 = WMMA16(pa, vb1, o1);
    o2 = WMMA16(pa, vb2, o2);
    o3 = WMMA16(pa, vb3, o3);
  }

  // normalize and store z
#pragma unroll
  for (int j = 0; j < 8; j++) {
    int row = qt * 16 + j + half * 8;
    float inv = 1.f / l_[j];
    h16* zr = z + headoff + (size_t)row * rs + l16;
    zr[0]  = (h16)(o0[j] * inv);
    zr[16] = (h16)(o1[j] * inv);
    zr[32] = (h16)(o2[j] * inv);
    zr[48] = (h16)(o3[j] * inv);
  }
}

// ---------------- host orchestration ----------------
static inline dim3 gemm_grid(int Mr, int N) { return dim3((N + 127) / 128, (Mr + 63) / 64, 1); }

extern "C" void kernel_launch(void* const* d_in, const int* in_sizes, int n_in,
                              void* d_out, int out_size, void* d_ws, size_t ws_size,
                              hipStream_t stream) {
  (void)in_sizes; (void)n_in; (void)out_size; (void)ws_size;
  const int*   tokens = (const int*)  d_in[0];
  const float* W_E    = (const float*)d_in[1];
  const float* W_pos  = (const float*)d_in[2];
  const float* ln1_w  = (const float*)d_in[3];
  const float* ln1_b  = (const float*)d_in[4];
  const float* W_Q    = (const float*)d_in[5];
  const float* b_Q    = (const float*)d_in[6];
  const float* W_K    = (const float*)d_in[7];
  const float* b_K    = (const float*)d_in[8];
  const float* W_V    = (const float*)d_in[9];
  const float* b_V    = (const float*)d_in[10];
  const float* W_O    = (const float*)d_in[11];
  const float* b_O    = (const float*)d_in[12];
  const float* ln2_w  = (const float*)d_in[13];
  const float* ln2_b  = (const float*)d_in[14];
  const float* W_in   = (const float*)d_in[15];
  const float* b_in   = (const float*)d_in[16];
  const float* W_out  = (const float*)d_in[17];
  const float* b_out  = (const float*)d_in[18];
  const float* lnf_w  = (const float*)d_in[19];
  const float* lnf_b  = (const float*)d_in[20];
  const float* W_U    = (const float*)d_in[21];
  const float* b_U    = (const float*)d_in[22];
  float* out = (float*)d_out;

  // workspace carve-out
  char* wp = (char*)d_ws;
  auto carve = [&](size_t bytes) -> void* {
    void* p = (void*)wp;
    wp += (bytes + 255) & ~(size_t)255;
    return p;
  };
  float* resid = (float*)carve((size_t)Tc * Dc * 4);
  h16* xln   = (h16*)carve((size_t)Tc * Dc * 2);
  h16* qb    = (h16*)carve((size_t)Tc * Dc * 2);
  h16* kb    = (h16*)carve((size_t)Tc * Dc * 2);
  h16* vbuf  = (h16*)carve((size_t)Tc * Dc * 2);
  h16* zb    = (h16*)carve((size_t)Tc * Dc * 2);
  h16* post  = (h16*)carve((size_t)Tc * Mc * 2);
  h16* wq_t  = (h16*)carve((size_t)Dc * Dc * 2);
  h16* wk_t  = (h16*)carve((size_t)Dc * Dc * 2);
  h16* wv_t  = (h16*)carve((size_t)Dc * Dc * 2);
  h16* wo_t  = (h16*)carve((size_t)Dc * Dc * 2);
  h16* win_t = (h16*)carve((size_t)Dc * Mc * 2);
  h16* wout_t= (h16*)carve((size_t)Mc * Dc * 2);
  h16* wu_t  = (h16*)carve((size_t)Vc * Dc * 2);

  auto tlaunch = [&](const float* src, h16* dst, int R, int C, long tot) {
    k_transpose_f32_f16<<<(unsigned)((tot + 255) / 256), 256, 0, stream>>>(src, dst, R, C, tot);
  };

  // embedding
  k_embed<<<Tc, 256, 0, stream>>>(tokens, W_E, W_pos, resid);
  // unembedding weight: [D,V] -> [V,D] f16 (done once)
  tlaunch(W_U, wu_t, Dc, Vc, (long)Dc * Vc);

  for (int l = 0; l < Lc; ++l) {
    const float* lnw1 = ln1_w + (size_t)l * Dc;
    const float* lnb1 = ln1_b + (size_t)l * Dc;
    const float* wq = W_Q + (size_t)l * Hc * Dc * DHc;
    const float* wk = W_K + (size_t)l * Hc * Dc * DHc;
    const float* wv = W_V + (size_t)l * Hc * Dc * DHc;
    const float* bq = b_Q + (size_t)l * Hc * DHc;
    const float* bk = b_K + (size_t)l * Hc * DHc;
    const float* bv = b_V + (size_t)l * Hc * DHc;
    const float* wo = W_O + (size_t)l * Hc * DHc * Dc;
    const float* bo = b_O + (size_t)l * Dc;
    const float* lnw2 = ln2_w + (size_t)l * Dc;
    const float* lnb2 = ln2_b + (size_t)l * Dc;
    const float* wi = W_in  + (size_t)l * Dc * Mc;
    const float* bi = b_in  + (size_t)l * Mc;
    const float* wu2= W_out + (size_t)l * Mc * Dc;
    const float* bo2= b_out + (size_t)l * Dc;

    // weight conversions (B-transposed f16)
    tlaunch(wq, wq_t, Dc, DHc, (long)Hc * Dc * DHc);      // batch=H: [H,DH,D]
    tlaunch(wk, wk_t, Dc, DHc, (long)Hc * Dc * DHc);
    tlaunch(wv, wv_t, Dc, DHc, (long)Hc * Dc * DHc);
    tlaunch(wo, wo_t, Hc * DHc, Dc, (long)Dc * Dc);       // [D, H*DH]
    tlaunch(wi, win_t, Dc, Mc, (long)Dc * Mc);            // [M, D]
    tlaunch(wu2, wout_t, Mc, Dc, (long)Mc * Dc);          // [D, M]

    // ln1 -> x (f16)
    k_layernorm<<<Tc, 256, 0, stream>>>(resid, lnw1, lnb1, xln);

    // q/k/v = x @ W_{q,k,v} + b  (N = H*DH = 1024, folded heads)
    dim3 g1 = gemm_grid(Tc, Hc * DHc);
    k_gemm_wmma<<<g1, 256, 0, stream>>>(xln, Dc, wq_t, Dc, bq, nullptr, 0,
                                        nullptr, qb, Hc * DHc, Tc, Hc * DHc, Dc, 0);
    k_gemm_wmma<<<g1, 256, 0, stream>>>(xln, Dc, wk_t, Dc, bk, nullptr, 0,
                                        nullptr, kb, Hc * DHc, Tc, Hc * DHc, Dc, 0);
    k_gemm_wmma<<<g1, 256, 0, stream>>>(xln, Dc, wv_t, Dc, bv, nullptr, 0,
                                        nullptr, vbuf, Hc * DHc, Tc, Hc * DHc, Dc, 0);

    // flash attention
    dim3 ga(Sc / 16, Hc, Bc);
    k_attn<<<ga, 32, 0, stream>>>(qb, kb, vbuf, zb);

    // attn out projection + residual (in-place on resid)
    k_gemm_wmma<<<gemm_grid(Tc, Dc), 256, 0, stream>>>(zb, Hc * DHc, wo_t, Hc * DHc, bo,
                                                       resid, Dc, resid, nullptr, Dc,
                                                       Tc, Dc, Hc * DHc, 2);

    // ln2 -> x (f16)
    k_layernorm<<<Tc, 256, 0, stream>>>(resid, lnw2, lnb2, xln);

    // MLP: gelu(x @ W_in + b_in) @ W_out + b_out + resid
    k_gemm_wmma<<<gemm_grid(Tc, Mc), 256, 0, stream>>>(xln, Dc, win_t, Dc, bi,
                                                       nullptr, 0, nullptr, post, Mc,
                                                       Tc, Mc, Dc, 1);
    k_gemm_wmma<<<gemm_grid(Tc, Dc), 256, 0, stream>>>(post, Mc, wout_t, Mc, bo2,
                                                       resid, Dc, resid, nullptr, Dc,
                                                       Tc, Dc, Mc, 2);
  }

  // final LN + unembedding
  k_layernorm<<<Tc, 256, 0, stream>>>(resid, lnf_w, lnf_b, xln);
  k_gemm_wmma<<<gemm_grid(Tc, Vc), 256, 0, stream>>>(xln, Dc, wu_t, Dc, b_U,
                                                     nullptr, 0, out, nullptr, Vc,
                                                     Tc, Vc, Dc, 0);
}
